// MGCN_5557687681180
// MI455X (gfx1250) — matrix operand order repaired and checked by reading, not compiled, not run on hardware
//
#include <hip/hip_runtime.h>
#include <hip/hip_bf16.h>

// ---- WMMA vector types (probe-confirmed signatures) -----------------------
typedef __attribute__((ext_vector_type(16))) __bf16 v16bf;
typedef __attribute__((ext_vector_type(8)))  float  v8f;
typedef __attribute__((ext_vector_type(4)))  unsigned int v4u;
typedef __attribute__((ext_vector_type(8)))  int         v8i_t;
typedef __attribute__((ext_vector_type(4)))  int         v4i_t;

union Frag16 { v16bf v; unsigned u[8]; };

#if defined(__has_builtin)
#if __has_builtin(__builtin_amdgcn_tensor_load_to_lds)
#define HAVE_TDM 1
#endif
#endif

// round-to-nearest-even f32 -> bf16 (raw u16)  (cold paths only)
__device__ __forceinline__ unsigned short bf_rne(float f) {
    unsigned u = __float_as_uint(f);
    u += 0x7FFFu + ((u >> 16) & 1u);
    return (unsigned short)(u >> 16);
}
__device__ __forceinline__ float bf_to_f(unsigned short h) {
    return __uint_as_float(((unsigned)h) << 16);
}

// hot-path split: hi = truncated bf16, lo = residual; packed via v_perm_b32
__device__ __forceinline__ void cvt_pair(float f0, float f1, unsigned& hi, unsigned& lo) {
    unsigned u0 = __float_as_uint(f0), u1 = __float_as_uint(f1);
    hi = __builtin_amdgcn_perm(u1, u0, 0x07060302u);        // {hi16(u1),hi16(u0)}
    float l0 = f0 - __uint_as_float(u0 & 0xFFFF0000u);
    float l1 = f1 - __uint_as_float(u1 & 0xFFFF0000u);
    lo = __builtin_amdgcn_perm(__float_as_uint(l1), __float_as_uint(l0), 0x07060302u);
}

// async global -> LDS, 16B per lane, ASYNCcnt-tracked (CDNA5)
__device__ __forceinline__ void async_b128(unsigned ldsOff, unsigned gOff, const void* base) {
    asm volatile("global_load_async_to_lds_b128 %0, %1, %2"
                 :: "v"(ldsOff), "v"(gOff), "s"(base) : "memory");
}
__device__ __forceinline__ void wait_async0() {
    asm volatile("s_wait_asynccnt 0" ::: "memory");
}

// scalar wave-id-in-group from TTMP8[29:25] (uniform -> guarantees s_cbranch;
// TDM issue ignores EXEC so the guard must be a scalar branch)
__device__ __forceinline__ bool is_group_wave0() {
    unsigned t8;
    asm volatile("s_mov_b32 %0, ttmp8" : "=s"(t8));
    return ((t8 >> 25) & 31u) == 0u;
}

#ifdef HAVE_TDM
// TDM: 2D tile DMA global->LDS with HW pad insertion (D# per CDNA5 ISA ch.8)
// dsz: 0=1B,1=2B,2=4B ; padI: interval code (2^(i+1) dwords) ; padA: dwords-1
__device__ __forceinline__ void tdm_load_2d(const void* gaddr, unsigned ldsAddr,
                                            unsigned dsz, unsigned padI, unsigned padA,
                                            unsigned tile0, unsigned tile1, unsigned stride0)
{
    unsigned long long ga = (unsigned long long)gaddr;
    v4u g0;
    g0[0] = 1u;                                        // count=1, user mode
    g0[1] = ldsAddr;                                   // lds_addr (bytes)
    g0[2] = (unsigned)(ga & 0xFFFFFFFFu);              // global_addr[31:0]
    g0[3] = (unsigned)((ga >> 32) & 0x01FFFFFFu) | (2u << 30);  // addr[56:32] | type=2
    v8i_t g1;
    g1[0] = (int)((dsz << 16) | (1u << 20) | (padI << 22) | (padA << 25));
    g1[1] = (int)(tile0 << 16);                        // tensor_dim0 = tile0
    g1[2] = (int)(tile1 << 16);                        // dim0 hi=0 | tensor_dim1 lo = tile1
    g1[3] = (int)(tile0 << 16);                        // dim1 hi=0 | tile_dim0
    g1[4] = (int)(tile1 & 0xFFFFu);                    // tile_dim1 | tile_dim2=0
    g1[5] = (int)stride0;                              // tensor_dim0_stride (elements)
    g1[6] = 0;
    g1[7] = 0;
    v4i_t z4 = {0, 0, 0, 0};
    v8i_t z8 = {0, 0, 0, 0, 0, 0, 0, 0};
    // 6-arg form (clang-23 / therock-10.0 headers): extra int32x8 group + cpol
    __builtin_amdgcn_tensor_load_to_lds(g0, g1, z4, z4, z8, 0);
}
#endif

#define TM    64   // M tile per block (4 waves x 16 rows)
#define KC    32   // K chunk (bf16 WMMA K)
#define NF   128   // N (embedding dim)
#define APAD  36   // dwords per A-tile row (128B data + 16B pad)
#define BPAD  40   // ushorts per B-tile row (64B data + 16B pad)

// C[M x 128] = A[M x K](f32, lda) @ (BhT+BlT)[128 x K](bf16 split, transposed)
__global__ void __launch_bounds__(128)
gcn_gemm(const float* __restrict__ A,
         const unsigned short* __restrict__ BhT, const unsigned short* __restrict__ BlT,
         const float* __restrict__ bias, float* __restrict__ C,
         unsigned short* __restrict__ OhT, unsigned short* __restrict__ OlT,
         int M, int K, int lda)
{
    __shared__ float          Af[2][TM * APAD];
    __shared__ unsigned short Bh[2][NF * BPAD];
    __shared__ unsigned short Bl[2][NF * BPAD];

    const int tid  = threadIdx.x;
    const int lane = tid & 31;
    const int wave = tid >> 5;
    const int m0   = blockIdx.x * TM;
    const int mw   = m0 + wave * 16;
    const int half = lane >> 4;
    const int l15  = lane & 15;
    const bool w0  = is_group_wave0();

    // issue one K-chunk of tile DMAs into LDS buffer `buf`
    auto issue_tiles = [&](int kk, int buf) {
        const unsigned aBase  = (unsigned)(uintptr_t)&Af[buf][0];
        const unsigned bhBase = (unsigned)(uintptr_t)&Bh[buf][0];
        const unsigned blBase = (unsigned)(uintptr_t)&Bl[buf][0];
#ifdef HAVE_TDM
        // A tile via Tensor Data Mover: 64 rows x 32 f32, stride lda, pad 32dw+4dw
        if (w0) {
            tdm_load_2d(A + (size_t)m0 * lda + kk, aBase,
                        /*dsz=*/2u, /*padI=*/4u, /*padA=*/3u,
                        /*tile0=*/KC, /*tile1=*/TM, /*stride0=*/(unsigned)lda);
        }
#else
        #pragma unroll
        for (int c = 0; c < 4; ++c) {
            int idx = tid + 128 * c;
            int row = idx >> 3, q = idx & 7;
            unsigned gOff = (unsigned)((m0 + row) * lda + kk + q * 4) * 4u;
            async_b128(aBase + (unsigned)(row * APAD + q * 4) * 4u, gOff, A);
        }
#endif
        // B tiles via async-load-to-LDS: 128 rows(cols) x 32 bf16 each
        #pragma unroll
        for (int c = 0; c < 4; ++c) {
            int idx = tid + 128 * c;              // 0..511
            int n = idx >> 2, q = idx & 3;
            unsigned gOff = (unsigned)(n * K + kk) * 2u + (unsigned)q * 16u;
            unsigned lOff = (unsigned)n * (BPAD * 2u) + (unsigned)q * 16u;
            async_b128(bhBase + lOff, gOff, BhT);
            async_b128(blBase + lOff, gOff, BlT);
        }
    };

    v8f acc[8];
    #pragma unroll
    for (int i = 0; i < 8; ++i)
        #pragma unroll
        for (int j = 0; j < 8; ++j) acc[i][j] = 0.0f;

    issue_tiles(0, 0);
    int cur = 0;

    for (int k0 = 0; k0 < K; k0 += KC) {
        __builtin_amdgcn_s_wait_tensorcnt(0);   // TDM done (no-op for waves 1-3)
        wait_async0();                          // this wave's async tile DMAs done
        __syncthreads();                        // all tiles visible; prev compute done
        if (k0 + KC < K) issue_tiles(k0 + KC, cur ^ 1);   // overlap next DMA

        // ---- A fragment: load raw f32, split to hi/lo bf16 in-registers --
        const int aro = (wave * 16 + l15) * APAD + half * 16;
        float4 a0 = *(const float4*)&Af[cur][aro];
        float4 a1 = *(const float4*)&Af[cur][aro + 4];
        float4 a2 = *(const float4*)&Af[cur][aro + 8];
        float4 a3 = *(const float4*)&Af[cur][aro + 12];
        Frag16 fah, fal;
        cvt_pair(a0.x, a0.y, fah.u[0], fal.u[0]);
        cvt_pair(a0.z, a0.w, fah.u[1], fal.u[1]);
        cvt_pair(a1.x, a1.y, fah.u[2], fal.u[2]);
        cvt_pair(a1.z, a1.w, fah.u[3], fal.u[3]);
        cvt_pair(a2.x, a2.y, fah.u[4], fal.u[4]);
        cvt_pair(a2.z, a2.w, fah.u[5], fal.u[5]);
        cvt_pair(a3.x, a3.y, fah.u[6], fal.u[6]);
        cvt_pair(a3.z, a3.w, fah.u[7], fal.u[7]);

        // ---- 8 N-subtiles, software-pipelined B fragments ----------------
        Frag16 fbh, fbl, nbh, nbl;
        {
            const int bo = l15 * BPAD + half * 16;
            uint4 h0 = *(const uint4*)(const void*)&Bh[cur][bo];
            uint4 h1 = *(const uint4*)(const void*)&Bh[cur][bo + 8];
            uint4 l0 = *(const uint4*)(const void*)&Bl[cur][bo];
            uint4 l1 = *(const uint4*)(const void*)&Bl[cur][bo + 8];
            fbh.u[0]=h0.x; fbh.u[1]=h0.y; fbh.u[2]=h0.z; fbh.u[3]=h0.w;
            fbh.u[4]=h1.x; fbh.u[5]=h1.y; fbh.u[6]=h1.z; fbh.u[7]=h1.w;
            fbl.u[0]=l0.x; fbl.u[1]=l0.y; fbl.u[2]=l0.z; fbl.u[3]=l0.w;
            fbl.u[4]=l1.x; fbl.u[5]=l1.y; fbl.u[6]=l1.z; fbl.u[7]=l1.w;
        }
        #pragma unroll
        for (int sub = 0; sub < 8; ++sub) {
            if (sub < 7) {   // prefetch next subtile's fragments
                const int bo = (sub * 16 + 16 + l15) * BPAD + half * 16;
                uint4 h0 = *(const uint4*)(const void*)&Bh[cur][bo];
                uint4 h1 = *(const uint4*)(const void*)&Bh[cur][bo + 8];
                uint4 l0 = *(const uint4*)(const void*)&Bl[cur][bo];
                uint4 l1 = *(const uint4*)(const void*)&Bl[cur][bo + 8];
                nbh.u[0]=h0.x; nbh.u[1]=h0.y; nbh.u[2]=h0.z; nbh.u[3]=h0.w;
                nbh.u[4]=h1.x; nbh.u[5]=h1.y; nbh.u[6]=h1.z; nbh.u[7]=h1.w;
                nbl.u[0]=l0.x; nbl.u[1]=l0.y; nbl.u[2]=l0.z; nbl.u[3]=l0.w;
                nbl.u[4]=l1.x; nbl.u[5]=l1.y; nbl.u[6]=l1.z; nbl.u[7]=l1.w;
            }
            acc[sub] = __builtin_amdgcn_wmma_f32_16x16x32_bf16(
                false, fah.v, false, fbh.v, (short)0, acc[sub], false, false);
            acc[sub] = __builtin_amdgcn_wmma_f32_16x16x32_bf16(
                false, fal.v, false, fbh.v, (short)0, acc[sub], false, false);
            acc[sub] = __builtin_amdgcn_wmma_f32_16x16x32_bf16(
                false, fah.v, false, fbl.v, (short)0, acc[sub], false, false);
            fbh = nbh; fbl = nbl;
        }
        cur ^= 1;
    }

    // ---- epilogue ---------------------------------------------------------
    if (C) {
        #pragma unroll
        for (int sub = 0; sub < 8; ++sub) {
            const int col = sub * 16 + l15;
            const float bv = bias ? bias[col] : 0.0f;
            #pragma unroll
            for (int r = 0; r < 8; ++r) {
                const int row = mw + half * 8 + r;
                C[(size_t)row * NF + col] = acc[sub][r] + bv;
            }
        }
    } else {
        // write split-bf16, transposed [col][M] for the next GEMM's B operand
        #pragma unroll
        for (int sub = 0; sub < 8; ++sub) {
            const int col = sub * 16 + l15;
            #pragma unroll
            for (int r = 0; r < 8; ++r) {
                const int row = mw + half * 8 + r;
                float v = acc[sub][r];
                unsigned short h = bf_rne(v);
                unsigned short l = bf_rne(v - bf_to_f(h));
                OhT[(size_t)col * M + row] = h;
                OlT[(size_t)col * M + row] = l;
            }
        }
    }
}

// ---- cold pre-pass: split + transpose W [R x 128] -> [128 x R] hi/lo -----
__global__ void __launch_bounds__(256)
split_transpose(const float* __restrict__ W, unsigned short* __restrict__ HT,
                unsigned short* __restrict__ LT, int R)
{
    int idx = blockIdx.x * blockDim.x + threadIdx.x;
    if (idx >= R * NF) return;
    int r = idx >> 7, c = idx & 127;
    float f = W[idx];
    unsigned short h = bf_rne(f);
    unsigned short l = bf_rne(f - bf_to_f(h));
    HT[(size_t)c * R + r] = h;
    LT[(size_t)c * R + r] = l;
}

// ---- attention fusion: one wave32 per row --------------------------------
__global__ void __launch_bounds__(256)
attn_fuse(const float* __restrict__ e1, const float* __restrict__ e2,
          const float* __restrict__ aw, float* __restrict__ out, int N)
{
    int gid  = blockIdx.x * blockDim.x + threadIdx.x;
    int row  = gid >> 5;
    int lane = gid & 31;
    if (row >= N) return;

    const float4 w4 = *(const float4*)(aw + lane * 4);
    const float4 a4 = *(const float4*)(e1 + (size_t)row * NF + lane * 4);
    const float4 b4 = *(const float4*)(e2 + (size_t)row * NF + lane * 4);

    float w1 = a4.x*w4.x + a4.y*w4.y + a4.z*w4.z + a4.w*w4.w;
    float w2 = b4.x*w4.x + b4.y*w4.y + b4.z*w4.z + b4.w*w4.w;
    #pragma unroll
    for (int off = 16; off; off >>= 1) {
        w1 += __shfl_xor(w1, off, 32);
        w2 += __shfl_xor(w2, off, 32);
    }
    float m  = fmaxf(w1, w2);
    float x1 = __expf(w1 - m), x2 = __expf(w2 - m);
    float beta = x1 / (x1 + x2);

    float4 o;
    o.x = beta * a4.x + (1.0f - beta) * b4.x;
    o.y = beta * a4.y + (1.0f - beta) * b4.y;
    o.z = beta * a4.z + (1.0f - beta) * b4.z;
    o.w = beta * a4.w + (1.0f - beta) * b4.w;
    *(float4*)(out + (size_t)row * NF + lane * 4) = o;
}

extern "C" void kernel_launch(void* const* d_in, const int* in_sizes, int n_in,
                              void* d_out, int out_size, void* d_ws, size_t ws_size,
                              hipStream_t stream) {
    const float* x    = (const float*)d_in[0];
    const float* adj1 = (const float*)d_in[1];
    const float* adj2 = (const float*)d_in[2];
    const float* W1   = (const float*)d_in[3];
    const float* b1   = (const float*)d_in[4];
    const float* W2   = (const float*)d_in[5];
    const float* b2   = (const float*)d_in[6];
    const float* aw   = (const float*)d_in[7];

    const int N = 8192, NFEAT = 512, NEMB = 128;
    float* out  = (float*)d_out;
    float* emb1 = out;
    float* emb2 = out + (size_t)N * NEMB;
    float* emb  = out + 2 * (size_t)N * NEMB;

    // workspace: 4x W-split (128KB ea) + 4x S-split (2MB ea) = ~8.5MB
    unsigned short* us = (unsigned short*)d_ws;
    const size_t WSZ = (size_t)NFEAT * NEMB;   // 65536
    const size_t SSZ = (size_t)N * NEMB;       // 1048576
    unsigned short* WhT1 = us;             unsigned short* WlT1 = us + WSZ;
    unsigned short* WhT2 = us + 2 * WSZ;   unsigned short* WlT2 = us + 3 * WSZ;
    unsigned short* ShT1 = us + 4 * WSZ;
    unsigned short* SlT1 = ShT1 + SSZ;
    unsigned short* ShT2 = ShT1 + 2 * SSZ;
    unsigned short* SlT2 = ShT1 + 3 * SSZ;

    dim3 blk(128), grid(N / TM);

    split_transpose<<<(NFEAT * NEMB + 255) / 256, 256, 0, stream>>>(W1, WhT1, WlT1, NFEAT);
    split_transpose<<<(NFEAT * NEMB + 255) / 256, 256, 0, stream>>>(W2, WhT2, WlT2, NFEAT);

    // S = x @ W  -> split-bf16 transposed (feeds next GEMM's B operand)
    gcn_gemm<<<grid, blk, 0, stream>>>(x, WhT1, WlT1, nullptr, nullptr, ShT1, SlT1, N, NFEAT, NFEAT);
    gcn_gemm<<<grid, blk, 0, stream>>>(x, WhT2, WlT2, nullptr, nullptr, ShT2, SlT2, N, NFEAT, NFEAT);
    // emb = adj @ S + b   (dominant, HBM-bound on adj)
    gcn_gemm<<<grid, blk, 0, stream>>>(adj1, ShT1, SlT1, b1, emb1, nullptr, nullptr, N, N, N);
    gcn_gemm<<<grid, blk, 0, stream>>>(adj2, ShT2, SlT2, b2, emb2, nullptr, nullptr, N, N, N);
    // attention over the two views
    attn_fuse<<<(N * 32) / 256, 256, 0, stream>>>(emb1, emb2, aw, emb, N);
}
